// Tylor_expansion_LlamaMLP_53961969107577
// MI455X (gfx1250) — matrix operand order repaired and checked by reading, not compiled, not run on hardware
//
#include <hip/hip_runtime.h>
#include <hip/hip_bf16.h>
#include <math.h>

// ---------------------------------------------------------------------------
// CDNA5 (gfx1250, wave32) bf16 WMMA GEMM engine, 4 fused launches:
//   MODE_TAYLOR : approx_up/gate_pre + Taylor-term epilogue -> Abig bf16 [4096,7168]
//   MODE_FUSE   : Abig @ concat(local_approx_output, fuse_weight[:,:,m])^T -> out (=)
//   MODE_SILU   : ffn gate/up + silu epilogue -> act bf16 [4096,9984]
//   MODE_DOWN   : act @ ffn_down^T -> out (+=)
// Core math: v_wmma_f32_16x16x32_bf16 (f32 accumulate).
// Data movement: double-buffered LDS pipeline; bf16-A tiles stream directly
// global->LDS via GLOBAL_LOAD_ASYNC_TO_LDS_B128 (ASYNCcnt).
// ---------------------------------------------------------------------------

typedef __attribute__((ext_vector_type(16))) __bf16 v16bf;
typedef __attribute__((ext_vector_type(8)))  float  v8f;

#define BM 64      // block tile M
#define BN 128     // block tile N
#define BK 32      // K step (one wmma K)
#define SK 40      // padded LDS K stride (80B rows, keeps 16B alignment)

enum { MODE_TAYLOR = 0, MODE_FUSE = 1, MODE_SILU = 2, MODE_DOWN = 3 };

union V16U { uint4 u[2]; v16bf v; };           // one wmma bf16 fragment (32B)
union V8B  { uint4 u;    __bf16 h[8];  };
union V16B { uint4 u[2]; __bf16 h[16]; };

__device__ __forceinline__ void wait_async0() {
#if __has_builtin(__builtin_amdgcn_s_wait_asynccnt)
    __builtin_amdgcn_s_wait_asynccnt(0);
#else
    asm volatile("s_wait_asynccnt 0x0" ::: "memory");
#endif
}

template <int MODE>
__global__ __launch_bounds__(256) void gemm_cdna5(
    const float*  __restrict__ Af,   // fp32 A (hidden_states) for TAYLOR/SILU
    const __bf16* __restrict__ Ab,   // bf16 A (workspace) for FUSE/DOWN
    const float*  __restrict__ W0,   // weight 0 [N,K] fp32
    const float*  __restrict__ W1,   // weight 1 [N,K] fp32 (dual modes)
    const float*  __restrict__ lao,  // local_approx_output [H,S] (FUSE)
    const float*  __restrict__ fw,   // fuse_weight [H,S,G]       (FUSE)
    const float*  __restrict__ lp,   // local_point [S]            (TAYLOR)
    const float*  __restrict__ df,   // discount_factor [G]        (TAYLOR)
    float*        __restrict__ outF, // fp32 out (FUSE '=', DOWN '+=')
    __bf16*       __restrict__ outB, // bf16 out (TAYLOR/SILU)
    int K, int ldOut)
{
    constexpr bool DUAL  = (MODE == MODE_TAYLOR || MODE == MODE_SILU);
    constexpr bool ABF16 = (MODE == MODE_FUSE   || MODE == MODE_DOWN);

    // double-buffered tiles (dual: 51.2 KB of the 320 KB WGP pool)
    __shared__ __align__(16) __bf16 As [2][BM][SK];
    __shared__ __align__(16) __bf16 Bs0[2][BN][SK];
    __shared__ __align__(16) __bf16 Bs1[DUAL ? 2 : 1][DUAL ? BN : 1][DUAL ? SK : 1];

    const int bM   = blockIdx.y * BM;
    const int bN   = blockIdx.x * BN;
    const int t    = threadIdx.x;
    const int lane = t & 31;
    const int wid  = t >> 5;          // 8 waves: 2 (M) x 4 (N)
    const int wm   = (wid & 1) * 32;
    const int wn   = (wid >> 1) * 32;
    const int tsub = lane & 15;       // row/col within 16x16 tile
    const int half = lane >> 4;       // lane group (ISA A/B fragment halves)

    // cooperative tile-loader indices (256 threads)
    const int ar = t >> 2, ac = (t & 3) * 8;   // A: 64 rows x 4 chunks of 8
    const int br = t >> 1, bc = (t & 1) * 16;  // B: 128 rows x 2 chunks of 16

    v8f acc0[2][2] = {};
    v8f acc1[2][2] = {};   // dual-B accumulators (gate / up)

    float aReg[8];          // staged fp32 A (fp32-A modes)
    float bReg0[16];        // staged fp32 B0
    float bReg1[16];        // staged fp32 B1 (dual modes)

    // ---- stage: issue global loads for tile at k0 (async A goes straight
    //      to LDS buffer `buf`; fp32 data lands in registers) ---------------
    auto stage = [&](int k0, int buf) {
        if constexpr (ABF16) {
            const __bf16* src = Ab + (size_t)(bM + ar) * K + (k0 + ac);
            const unsigned ldsoff = (unsigned)(uintptr_t)&As[buf][ar][ac];
            asm volatile("global_load_async_to_lds_b128 %0, %1, off"
                         :: "v"(ldsoff),
                            "v"((unsigned long long)(uintptr_t)src)
                         : "memory");
            __builtin_prefetch(src + BK, 0, 0);
        } else {
            const float* src = Af + (size_t)(bM + ar) * K + (k0 + ac);
            float4 f0 = ((const float4*)src)[0];
            float4 f1 = ((const float4*)src)[1];
            aReg[0]=f0.x; aReg[1]=f0.y; aReg[2]=f0.z; aReg[3]=f0.w;
            aReg[4]=f1.x; aReg[5]=f1.y; aReg[6]=f1.z; aReg[7]=f1.w;
            __builtin_prefetch(src + BK, 0, 0);
        }
        if constexpr (MODE == MODE_FUSE) {
            const int nrow = bN + br;                       // h index
#pragma unroll
            for (int j = 0; j < 16; ++j) {
                const int k = k0 + bc + j;
                bReg0[j] = (k < 1024)
                    ? lao[(size_t)nrow * 1024 + k]
                    : fw[((size_t)nrow * 1024 + (k & 1023)) * 6 + ((k >> 10) - 1)];
            }
        } else {
            const float* s0 = W0 + (size_t)(bN + br) * K + (k0 + bc);
#pragma unroll
            for (int j = 0; j < 4; ++j) {
                float4 f = ((const float4*)s0)[j];
                bReg0[j*4+0]=f.x; bReg0[j*4+1]=f.y;
                bReg0[j*4+2]=f.z; bReg0[j*4+3]=f.w;
            }
            __builtin_prefetch(s0 + BK, 0, 0);
            if constexpr (DUAL) {
                const float* s1 = W1 + (size_t)(bN + br) * K + (k0 + bc);
#pragma unroll
                for (int j = 0; j < 4; ++j) {
                    float4 f = ((const float4*)s1)[j];
                    bReg1[j*4+0]=f.x; bReg1[j*4+1]=f.y;
                    bReg1[j*4+2]=f.z; bReg1[j*4+3]=f.w;
                }
            }
        }
    };

    // ---- commit: convert staged registers to bf16, store into LDS buffer;
    //      fence the async A copy -------------------------------------------
    auto commit = [&](int buf) {
        if constexpr (!ABF16) {
            V8B pk;
#pragma unroll
            for (int j = 0; j < 8; ++j) pk.h[j] = (__bf16)aReg[j];
            *(uint4*)&As[buf][ar][ac] = pk.u;
        }
        {
            V16B pk;
#pragma unroll
            for (int j = 0; j < 16; ++j) pk.h[j] = (__bf16)bReg0[j];
            *(uint4*)&Bs0[buf][br][bc]     = pk.u[0];
            *(uint4*)&Bs0[buf][br][bc + 8] = pk.u[1];
        }
        if constexpr (DUAL) {
            V16B pk;
#pragma unroll
            for (int j = 0; j < 16; ++j) pk.h[j] = (__bf16)bReg1[j];
            *(uint4*)&Bs1[buf][br][bc]     = pk.u[0];
            *(uint4*)&Bs1[buf][br][bc + 8] = pk.u[1];
        }
        if constexpr (ABF16) wait_async0();
    };

    // ---- compute: build contiguous-k fragments from LDS, run WMMAs ----------
    auto compute = [&](int buf) {
        // A frag (16-bit 16x32): lanes 0-15 K{0..15}, lanes 16-31
        // K{16..31} -> two ds_load_b128 each.
        V16U a[2], b0[2], b1[2];
#pragma unroll
        for (int ti = 0; ti < 2; ++ti) {
            const int m = wm + ti * 16 + tsub;
            a[ti].u[0] = *(const uint4*)&As[buf][m][half * 16];
            a[ti].u[1] = *(const uint4*)&As[buf][m][half * 16 + 8];
        }
        // B frag: lanes 0-15 K=0..15, lanes 16-31 K=16..31 (Bs holds B^T).
#pragma unroll
        for (int tj = 0; tj < 2; ++tj) {
            const int n  = wn + tj * 16 + tsub;
            const int kb = half * 16;
            b0[tj].u[0] = *(const uint4*)&Bs0[buf][n][kb];
            b0[tj].u[1] = *(const uint4*)&Bs0[buf][n][kb + 8];
            if constexpr (DUAL) {
                b1[tj].u[0] = *(const uint4*)&Bs1[buf][n][kb];
                b1[tj].u[1] = *(const uint4*)&Bs1[buf][n][kb + 8];
            }
        }
#pragma unroll
        for (int ti = 0; ti < 2; ++ti)
#pragma unroll
            for (int tj = 0; tj < 2; ++tj) {
                acc0[ti][tj] = __builtin_amdgcn_wmma_f32_16x16x32_bf16(
                    false, a[ti].v, false, b0[tj].v, (short)0, acc0[ti][tj],
                    false, false);
                if constexpr (DUAL)
                    acc1[ti][tj] = __builtin_amdgcn_wmma_f32_16x16x32_bf16(
                        false, a[ti].v, false, b1[tj].v, (short)0, acc1[ti][tj],
                        false, false);
            }
    };

    // ---- double-buffered main pipeline ------------------------------------
    const int nk = K / BK;
    stage(0, 0);
    commit(0);
    __syncthreads();
    for (int i = 0; i < nk; ++i) {
        const int  cur  = i & 1;
        const bool more = (i + 1) < nk;
        if (more) stage((i + 1) * BK, cur ^ 1);   // overlap with compute below
        compute(cur);
        if (more) commit(cur ^ 1);
        __syncthreads();
    }

    // ---- epilogue (C/D layout: row = half*8+r, col = tsub) ----------------
    float dfv[6] = {};
    float lpv[2] = {};
    if constexpr (MODE == MODE_TAYLOR) {
#pragma unroll
        for (int m = 0; m < 6; ++m) dfv[m] = df[m];
#pragma unroll
        for (int tj = 0; tj < 2; ++tj) lpv[tj] = lp[bN + wn + tj * 16 + tsub];
    }
#pragma unroll
    for (int ti = 0; ti < 2; ++ti)
#pragma unroll
        for (int tj = 0; tj < 2; ++tj)
#pragma unroll
            for (int r = 0; r < 8; ++r) {
                const int row = bM + wm + ti * 16 + half * 8 + r;
                const int col = bN + wn + tj * 16 + tsub;
                const float v0 = acc0[ti][tj][r];
                if constexpr (MODE == MODE_TAYLOR) {
                    // fp32 Taylor terms: dterm = |d|^m/m! (log-space), sign on
                    // odd m, trust-region cutoff for m>4.
                    const float sg    = acc1[ti][tj][r];
                    const float delta = sg - lpv[tj];
                    const float neg   = delta > 0.f ? 1.f : -1.f;
                    const float dabs  = fabsf(delta);
                    const float ldl   = __logf(dabs);
                    const float keep  = dabs <= 2.5f ? 1.f : 0.f;
                    __bf16* dst = outB + (size_t)row * ldOut + col;
                    dst[0] = (__bf16)v0;
#pragma unroll
                    for (int m = 1; m <= 6; ++m) {
                        float dt = __expf(ldl * (float)m - dfv[m - 1]);
                        if (m & 1) dt *= neg;
                        if (m > 4) dt *= keep;
                        dst[(size_t)m * 1024] = (__bf16)(dt * v0);
                    }
                } else if constexpr (MODE == MODE_SILU) {
                    const float u   = acc1[ti][tj][r];
                    const float act = (v0 / (1.f + __expf(-v0))) * u;
                    outB[(size_t)row * ldOut + col] = (__bf16)act;
                } else if constexpr (MODE == MODE_FUSE) {
                    outF[(size_t)row * ldOut + col] = v0;
                } else { // MODE_DOWN
                    outF[(size_t)row * ldOut + col] += v0;
                }
            }
}

// ---------------------------------------------------------------------------
extern "C" void kernel_launch(void* const* d_in, const int* in_sizes, int n_in,
                              void* d_out, int out_size, void* d_ws, size_t ws_size,
                              hipStream_t stream) {
    (void)in_sizes; (void)n_in; (void)out_size; (void)ws_size;
    const float* hidden = (const float*)d_in[0];   // [2,2048,4096]
    const float* up_w   = (const float*)d_in[1];   // [1024,4096]
    const float* gate_w = (const float*)d_in[2];   // [1024,4096]
    const float* lp     = (const float*)d_in[3];   // [1024]
    const float* lao    = (const float*)d_in[4];   // [4096,1024]
    const float* fw     = (const float*)d_in[5];   // [4096,1024,6]
    const float* df     = (const float*)d_in[6];   // [6]
    const float* gw     = (const float*)d_in[7];   // [9984,4096]
    const float* uw     = (const float*)d_in[8];   // [9984,4096]
    const float* dw     = (const float*)d_in[9];   // [4096,9984]
    float* out = (float*)d_out;                    // [2,2048,4096]

    const int M = 4096; // B*T
    __bf16* Abig = (__bf16*)d_ws;                  // [4096,7168] = 58.7 MB
    __bf16* act  = Abig + (size_t)M * 7168;        // [4096,9984] = 81.8 MB

    dim3 blk(256);
    // 1) approx_up & gate_pre (shared A tiles) + Taylor epilogue -> Abig
    gemm_cdna5<MODE_TAYLOR><<<dim3(1024 / BN, M / BM), blk, 0, stream>>>(
        hidden, nullptr, up_w, gate_w, nullptr, nullptr, lp, df,
        nullptr, Abig, 4096, 7168);
    // 2) K=7168 fused Taylor-sum GEMM -> out (initializes d_out)
    gemm_cdna5<MODE_FUSE><<<dim3(4096 / BN, M / BM), blk, 0, stream>>>(
        nullptr, Abig, nullptr, nullptr, lao, fw, nullptr, nullptr,
        out, nullptr, 7168, 4096);
    // 3) ffn gate/up (shared A tiles) + SiLU epilogue -> act
    gemm_cdna5<MODE_SILU><<<dim3(9984 / BN, M / BM), blk, 0, stream>>>(
        hidden, nullptr, gw, uw, nullptr, nullptr, nullptr, nullptr,
        nullptr, act, 4096, 9984);
    // 4) out += act @ ffn_down^T
    gemm_cdna5<MODE_DOWN><<<dim3(4096 / BN, M / BM), blk, 0, stream>>>(
        nullptr, act, dw, nullptr, nullptr, nullptr, nullptr, nullptr,
        out, nullptr, 9984, 4096);
}
